// mean_aggre_6631429505472
// MI455X (gfx1250) — matrix operand order repaired and checked by reading, not compiled, not run on hardware
//
#include <hip/hip_runtime.h>

// ---------------------------------------------------------------------------
// Fused relu(X @ W^T + b) + segment-mean scatter for MI455X (gfx1250).
// GEMM on the f16 WMMA pipe via fp16x3 error-split (x_hi*w_hi + x_hi*w_lo +
// x_lo*w_hi, fp32 accumulate) -> ~fp32 accuracy. W is pre-split once into
// fp16 hi/lo planes (workspace) so the hot kernel does zero W conversion.
// Scatter uses no-return global_atomic_add_f32 into an L2-resident 20.5MB
// accumulator (192MB L2), then a finalize kernel divides by counts.
// ---------------------------------------------------------------------------

typedef _Float16 v4h  __attribute__((ext_vector_type(4)));
typedef _Float16 v8h  __attribute__((ext_vector_type(8)));
typedef _Float16 v16h __attribute__((ext_vector_type(16)));
typedef float    v4f  __attribute__((ext_vector_type(4)));
typedef float    v8f  __attribute__((ext_vector_type(8)));

#define D        128
#define TILE     16
#define EPB      64               // edges per block
#define MT       (EPB / TILE)     // 4 M-tiles
#define NWAVES   8                // 8 waves x 16 features = 128 = D
#define THREADS  (NWAVES * 32)
#define XSTRIDE  136              // half-elem row stride: 272B (16B aligned, bank skew)

__global__ __launch_bounds__(THREADS)
void zero_ws_kernel(float* __restrict__ p, int n) {
    int i = blockIdx.x * blockDim.x + threadIdx.x;
    if (i < n) p[i] = 0.0f;
}

// one-shot: split W (fp32) into fp16 hi/lo planes
__global__ __launch_bounds__(256)
void wsplit_kernel(const float* __restrict__ Wm,
                   _Float16* __restrict__ whi,
                   _Float16* __restrict__ wlo, int nElem) {
    int i = blockIdx.x * blockDim.x + threadIdx.x;
    if (i < nElem) {
        float w = Wm[i];
        _Float16 h = (_Float16)w;
        whi[i] = h;
        wlo[i] = (_Float16)(w - (float)h);
    }
}

__global__ __launch_bounds__(THREADS)
void msg_gemm_scatter_kernel(const float*    __restrict__ x,      // [E, D]
                             const int*      __restrict__ dst,    // [E]
                             const _Float16* __restrict__ whi,    // [D, D] fp16 hi
                             const _Float16* __restrict__ wlo,    // [D, D] fp16 lo
                             const float*    __restrict__ bias,   // [D]
                             float* __restrict__ sums,            // [N, D]
                             float* __restrict__ counts,          // [N]
                             int E) {
    __shared__ _Float16 lds_xhi[EPB * XSTRIDE];
    __shared__ _Float16 lds_xlo[EPB * XSTRIDE];
    __shared__ int      lds_dst[EPB];

    const int t    = threadIdx.x;
    const int wave = t >> 5;
    const int lane = t & 31;
    const int n    = lane & 15;    // B/C column; A row within M-tile
    const int hi   = lane >> 4;    // half-wave -> K-chunk selector
    const int eb   = blockIdx.x * EPB;

    // ---- stage 64x128 fp32 tile, splitting into fp16 hi/lo halves in LDS ----
    {
        const v4f* xg = (const v4f*)(x + (size_t)eb * D);
        #pragma unroll
        for (int i = 0; i < (EPB * D / 4) / THREADS; ++i) {
            int flat = t + i * THREADS;      // float4 index in tile
            int row  = flat >> 5;            // 32 float4 per 128-float row
            int c4   = flat & 31;
            v4f v = {0.0f, 0.0f, 0.0f, 0.0f};
            if (eb + row < E) v = xg[(size_t)row * 32 + c4];
            v4h vh, vl;
            vh[0] = (_Float16)v.x;  vl[0] = (_Float16)(v.x - (float)vh[0]);
            vh[1] = (_Float16)v.y;  vl[1] = (_Float16)(v.y - (float)vh[1]);
            vh[2] = (_Float16)v.z;  vl[2] = (_Float16)(v.z - (float)vh[2]);
            vh[3] = (_Float16)v.w;  vl[3] = (_Float16)(v.w - (float)vh[3]);
            *(v4h*)&lds_xhi[row * XSTRIDE + c4 * 4] = vh;
            *(v4h*)&lds_xlo[row * XSTRIDE + c4 * 4] = vl;
        }
        if (t < EPB) lds_dst[t] = (eb + t < E) ? dst[eb + t] : 0;
    }

    // ---- load pre-split B fragments (fp16, contiguous 16B chunks) ----
    // B[k][nn] = W[featBase+nn][k]; lane holds K = ks*32 + 16*hi .. +15.
    const int featBase = wave * TILE;
    const _Float16* whrow = whi + (size_t)(featBase + n) * D;
    const _Float16* wlrow = wlo + (size_t)(featBase + n) * D;
    v16h bhi[4], blo[4];
    #pragma unroll
    for (int ks = 0; ks < 4; ++ks) {
        const int base = ks * 32 + 16 * hi;
        union { v16h v; v8h h[2]; } bh, bl;
        bh.h[0] = *(const v8h*)(whrow + base);       // global_load_b128
        bh.h[1] = *(const v8h*)(whrow + base + 8);
        bl.h[0] = *(const v8h*)(wlrow + base);
        bl.h[1] = *(const v8h*)(wlrow + base + 8);
        bhi[ks] = bh.v;
        blo[ks] = bl.v;
    }
    const float bv = bias[featBase + n];

    __syncthreads();

    // ---- 4 M-tiles; K=128 in 4 steps of 32; 3 WMMAs per step (fp16x3) ----
    #pragma unroll
    for (int mt = 0; mt < MT; ++mt) {
        v8f c = {0.f, 0.f, 0.f, 0.f, 0.f, 0.f, 0.f, 0.f};
        const _Float16* arow_h = &lds_xhi[(mt * TILE + n) * XSTRIDE];
        const _Float16* arow_l = &lds_xlo[(mt * TILE + n) * XSTRIDE];
        #pragma unroll
        for (int ks = 0; ks < 4; ++ks) {
            // A layout (16-bit, 16x32): v0..7 <- K=16hi+0..15 (matches B)
            const int kb = ks * 32 + 16 * hi;
            union { v16h v; v8h h[2]; } ah, al;
            ah.h[0] = *(const v8h*)(arow_h + kb);        // ds_load_b128
            ah.h[1] = *(const v8h*)(arow_h + kb + 8);
            al.h[0] = *(const v8h*)(arow_l + kb);
            al.h[1] = *(const v8h*)(arow_l + kb + 8);
            c = __builtin_amdgcn_wmma_f32_16x16x32_f16(false, ah.v, false, bhi[ks],
                                                       (short)0, c, false, false);
            c = __builtin_amdgcn_wmma_f32_16x16x32_f16(false, ah.v, false, blo[ks],
                                                       (short)0, c, false, false);
            c = __builtin_amdgcn_wmma_f32_16x16x32_f16(false, al.v, false, bhi[ks],
                                                       (short)0, c, false, false);
        }
        // ---- fused bias + ReLU + scatter-add ----
        #pragma unroll
        for (int r = 0; r < 8; ++r) {
            int m = r + 8 * hi;                // C layout: vgpr r -> M = r + 8*hi
            int e = eb + mt * TILE + m;
            if (e < E) {
                int   dn = lds_dst[mt * TILE + m];
                float v  = c[r] + bv;
                v = v > 0.0f ? v : 0.0f;
                atomicAdd(&sums[(size_t)dn * D + featBase + n], v);  // no-return
            }
        }
    }

    if (t < EPB && (eb + t) < E) atomicAdd(&counts[lds_dst[t]], 1.0f);
}

__global__ __launch_bounds__(256)
void finalize_kernel(const float* __restrict__ sums,
                     const float* __restrict__ counts,
                     float* __restrict__ out, int total4) {
    int i = blockIdx.x * blockDim.x + threadIdx.x;   // float4 index
    if (i < total4) {
        int   node = i >> 5;                          // 32 float4 per node row
        float cnt  = counts[node];
        cnt = cnt > 1.0f ? cnt : 1.0f;
        float inv = 1.0f / cnt;
        v4f s = ((const v4f*)sums)[i];
        v4f o = {s.x * inv, s.y * inv, s.z * inv, s.w * inv};
        ((v4f*)out)[i] = o;
    }
}

extern "C" void kernel_launch(void* const* d_in, const int* in_sizes, int n_in,
                              void* d_out, int out_size, void* d_ws, size_t ws_size,
                              hipStream_t stream) {
    const float* x    = (const float*)d_in[0];   // src_emb [E, D]
    const int*   dst  = (const int*)  d_in[1];   // dst_idx [E]
    const float* Wm   = (const float*)d_in[2];   // W [D, D]
    const float* bias = (const float*)d_in[3];   // b [D]

    const int E  = in_sizes[0] / D;
    const int Nn = out_size / D;

    // workspace layout: sums [Nn*D] f32 | counts [Nn] f32 | whi, wlo [D*D] f16
    float* sums   = (float*)d_ws;
    float* counts = sums + (size_t)Nn * D;
    size_t offF   = (((size_t)Nn * D + (size_t)Nn) + 7) & ~(size_t)7; // 32B align
    _Float16* whi = (_Float16*)(sums + offF);
    _Float16* wlo = whi + (size_t)D * D;

    int zn = Nn * D + Nn;
    zero_ws_kernel<<<(zn + THREADS - 1) / THREADS, THREADS, 0, stream>>>(sums, zn);

    wsplit_kernel<<<(D * D + 255) / 256, 256, 0, stream>>>(Wm, whi, wlo, D * D);

    int nblocks = (E + EPB - 1) / EPB;
    msg_gemm_scatter_kernel<<<nblocks, THREADS, 0, stream>>>(x, dst, whi, wlo, bias,
                                                             sums, counts, E);

    int t4 = (Nn * D) / 4;
    finalize_kernel<<<(t4 + 255) / 256, 256, 0, stream>>>(sums, counts,
                                                          (float*)d_out, t4);
}
